// NeuralParametrizedBSDF2_76828374991342
// MI455X (gfx1250) — matrix operand order repaired and checked by reading, compiled, not run on hardware
//
#include <hip/hip_runtime.h>
#include <math.h>

// ---------------------------------------------------------------------------
// Types
// ---------------------------------------------------------------------------
typedef _Float16 v16h  __attribute__((ext_vector_type(16)));
typedef float    v8f   __attribute__((ext_vector_type(8)));
typedef _Float16 half8 __attribute__((ext_vector_type(8)));

// ---------------------------------------------------------------------------
// Layer tables (shared by prep kernel, main kernel, host sizing)
//
// Padded weight storage: each layer = rowTiles tiles of 16 rows; every row has
// stride 72 halves (144 B) -> bank-friendly b128 LDS loads; row data holds
// kChunks*32 halves (K padded to 32).  Activation rows use the same 72-half
// stride.  Activation slot layout per point (uv chain):
//   [ c features 0..26 | pad 27..31 | hidden x 32..47 | zeros 48..63 ]
// ang chain uses slots 0..63 directly (input 0..10, hidden 0..63).
// ---------------------------------------------------------------------------
struct LayerT {
  int rows, rowTiles, kChunks, bStart, split, mlp, relu, outBase;
  int wSrc, cols, bSrc, wDst, bDst;
};

struct Tables {
  LayerT L[32];
  int wHalves, bFloats;
  constexpr Tables() : L(), wHalves(0), bFloats(0) {
    int wDst = 0, bDst = 0, wSrc = 0, bSrc = 0;
    // uv MLP: Ws[0..25]; hidden index i=k-1 for k in 1..24; skip at i=3,6,...,21
    for (int k = 0; k < 26; ++k) {
      int rows = (k == 25) ? 3 : 16;
      int fan  = (k == 0) ? 27 : (k == 25) ? 16
               : ((((k - 1) % 3) == 0 && (k - 1) > 0 && (k - 1) != 23) ? 43 : 16);
      int split = (fan == 43) ? 27 : 0;
      int kc    = split ? 2 : 1;
      int bSt   = (k == 0 || split) ? 0 : 1;
      int rt    = (rows + 15) / 16;
      L[k] = LayerT{rows, rt, kc, bSt, split, 0, (k >= 1 && k <= 24) ? 1 : 0, 32,
                    wSrc, fan, bSrc, wDst, bDst};
      wSrc += rows * fan; bSrc += rows; wDst += rt * 16 * 72; bDst += rt * 16;
    }
    // ang MLP: Ws[0..5], hidden 64, no skip (i=3 is last hidden layer)
    wSrc = 0; bSrc = 0;
    for (int k = 0; k < 6; ++k) {
      int rows = (k == 5) ? 3 : 64;
      int fan  = (k == 0) ? 11 : 64;
      int kc   = (fan > 32) ? 2 : 1;
      int rt   = (rows + 15) / 16;
      L[26 + k] = LayerT{rows, rt, kc, 0, 0, 1, (k >= 1 && k <= 4) ? 1 : 0, 0,
                         wSrc, fan, bSrc, wDst, bDst};
      wSrc += rows * fan; bSrc += rows; wDst += rt * 16 * 72; bDst += rt * 16;
    }
    wHalves = wDst; bFloats = bDst;
  }
};

static constexpr Tables kT{};

static constexpr int W_BYTES   = kT.wHalves * 2;          // 108288
static constexpr int B_BYTES   = kT.bFloats * 4;          // 3008
static constexpr int STAGE_B   = W_BYTES + B_BYTES;       // 111296 (16B aligned)
static constexpr int ACT_OFF   = STAGE_B;
static constexpr int ROW_B     = 144;                     // 72 halves
static constexpr int ACT_BUF   = 16 * ROW_B;              // 2304 B per buffer
static constexpr int WAVES     = 8;                       // 256 threads / wave32
static constexpr int SMEM_B    = ACT_OFF + WAVES * 2 * ACT_BUF; // 148160 B

// ---------------------------------------------------------------------------
// Prep kernel: f32 weights/biases -> padded f16 weight image + f32 bias image
// in d_ws (deterministic, rewrites every element each call).
// ---------------------------------------------------------------------------
__global__ void bsdf_prep(const float* __restrict__ uvW, const float* __restrict__ uvB,
                          const float* __restrict__ angW, const float* __restrict__ angB,
                          _Float16* __restrict__ wOut, float* __restrict__ bOut) {
  int tid = blockIdx.x * blockDim.x + threadIdx.x;
  int stride = gridDim.x * blockDim.x;
  for (int idx = tid; idx < kT.wHalves; idx += stride) {
    int li = 0;
    while (li + 1 < 32 && idx >= kT.L[li + 1].wDst) ++li;
    const LayerT Ld = kT.L[li];
    int local = idx - Ld.wDst;
    int tile  = local / 1152;
    int rr    = (local % 1152) / 72;
    int k     = local % 72;
    int row   = tile * 16 + rr;
    float v = 0.f;
    if (row < Ld.rows && k < Ld.kChunks * 32) {
      int c = -1;
      if (Ld.split) {                       // [orig 0..26 | pad | x part at 32..]
        if (k < Ld.split) c = k;
        else if (k >= 32 && (k - 32) < (Ld.cols - Ld.split)) c = Ld.split + (k - 32);
      } else if (k < Ld.cols) c = k;
      if (c >= 0) {
        const float* src = Ld.mlp ? angW : uvW;
        v = src[Ld.wSrc + row * Ld.cols + c];
      }
    }
    wOut[idx] = (_Float16)v;
  }
  for (int idx = tid; idx < kT.bFloats; idx += stride) {
    int li = 0;
    while (li + 1 < 32 && idx >= kT.L[li + 1].bDst) ++li;
    const LayerT Ld = kT.L[li];
    int r = idx - Ld.bDst;
    bOut[idx] = (r < Ld.rows) ? (Ld.mlp ? angB : uvB)[Ld.bSrc + r] : 0.f;
  }
}

// ---------------------------------------------------------------------------
// Device helpers
// ---------------------------------------------------------------------------
struct F3 { float x, y, z; };
__device__ __forceinline__ F3 nrm3(F3 a) {
  float l = sqrtf(a.x * a.x + a.y * a.y + a.z * a.z);
  float r = 1.f / fmaxf(l, 1e-12f);
  return F3{a.x * r, a.y * r, a.z * r};
}

// One hidden MLP layer: D(16xN16) = Wtile(16x32) * Act(32x16) per K-chunk via
// WMMA, then bias (+ optional ReLU), f16-pack, store to dst activation tile.
// ln = lane%16 (point / row-in-tile), lh = lane/16 (K half selector).
// Called only with compile-time-constant Ld (unrolled chains) so all fields
// fold to immediates and the t/j loops fully unroll.
__device__ __forceinline__ void run_layer(const LayerT Ld, const char* __restrict__ smemBase,
                                          const char* __restrict__ src, char* __restrict__ dst,
                                          int ln, int lh) {
  const float* sb = (const float*)(smemBase + W_BYTES);
#pragma unroll
  for (int t = 0; t < Ld.rowTiles; ++t) {
    v8f acc = {0.f, 0.f, 0.f, 0.f, 0.f, 0.f, 0.f, 0.f};
    const char* wt = smemBase + (size_t)(Ld.wDst + t * 1152) * 2 + ln * ROW_B + lh * 16;
    const char* bb = src + ln * ROW_B + Ld.bStart * 64 + lh * 16;
#pragma unroll
    for (int j = 0; j < Ld.kChunks; ++j) {
      union { v16h v; half8 h[2]; } A, B;
      A.h[0] = *(const half8*)(wt + j * 64);        // K = 8*lh .. +7
      A.h[1] = *(const half8*)(wt + j * 64 + 32);   // K = 16+8*lh .. +7
      B.h[0] = *(const half8*)(bb + j * 64);        // K = 8*lh .. +7
      B.h[1] = *(const half8*)(bb + j * 64 + 32);   // K = 16+8*lh .. +7
      acc = __builtin_amdgcn_wmma_f32_16x16x32_f16(false, A.v, false, B.v,
                                                   (short)0, acc, false, false);
    }
    const float* bp = sb + Ld.bDst + t * 16 + lh * 8;
    half8 hv;
#pragma unroll
    for (int r = 0; r < 8; ++r) {
      float yy = acc[r] + bp[r];
      if (Ld.relu) yy = fmaxf(yy, 0.f);
      hv[r] = (_Float16)yy;
    }
    *(half8*)(dst + ln * ROW_B + (size_t)(Ld.outBase + t * 16 + lh * 8) * 2) = hv;
  }
}

// Final (3-row) layer: keep result in registers (rows 0..2, lanes 0..15).
__device__ __forceinline__ void run_final(const LayerT Ld, const char* __restrict__ smemBase,
                                          const char* __restrict__ src,
                                          int ln, int lh, float* y3) {
  const float* sb = (const float*)(smemBase + W_BYTES);
  v8f acc = {0.f, 0.f, 0.f, 0.f, 0.f, 0.f, 0.f, 0.f};
  const char* wt = smemBase + (size_t)Ld.wDst * 2 + ln * ROW_B + lh * 16;
  const char* bb = src + ln * ROW_B + Ld.bStart * 64 + lh * 16;
#pragma unroll
  for (int j = 0; j < Ld.kChunks; ++j) {
    union { v16h v; half8 h[2]; } A, B;
    A.h[0] = *(const half8*)(wt + j * 64);
    A.h[1] = *(const half8*)(wt + j * 64 + 32);
    B.h[0] = *(const half8*)(bb + j * 64);
    B.h[1] = *(const half8*)(bb + j * 64 + 32);
    acc = __builtin_amdgcn_wmma_f32_16x16x32_f16(false, A.v, false, B.v,
                                                 (short)0, acc, false, false);
  }
  const float* bp = sb + Ld.bDst + lh * 8;
  y3[0] = acc[0] + bp[0];
  y3[1] = acc[1] + bp[1];
  y3[2] = acc[2] + bp[2];
}

// ---------------------------------------------------------------------------
// Main kernel
// ---------------------------------------------------------------------------
__global__ __launch_bounds__(256)
void bsdf_main(const float* __restrict__ x, const uint4* __restrict__ stage,
               const float* __restrict__ bu, const float* __restrict__ bfi,
               const float* __restrict__ ban, float* __restrict__ out,
               int nPoints, int nTiles) {
  extern __shared__ __align__(16) char smem[];

  // Stage weights + biases into LDS (once per block).
  {
    uint4* s4 = (uint4*)smem;
    for (int i = threadIdx.x; i < STAGE_B / 16; i += blockDim.x) s4[i] = stage[i];
  }
  __syncthreads();

  const int wv = threadIdx.x >> 5;
  const int lane = threadIdx.x & 31;
  const int ln = lane & 15;      // point within tile / weight row
  const int lh = lane >> 4;      // K half selector
  char* act0 = smem + ACT_OFF + wv * 2 * ACT_BUF;
  char* act1 = act0 + ACT_BUF;

  // Fourier bases -> (wave-uniform) registers.
  float BA[12], BU[16], BF[4];
#pragma unroll
  for (int i = 0; i < 12; ++i) BA[i] = ban[i];
#pragma unroll
  for (int i = 0; i < 16; ++i) BU[i] = bu[i];
#pragma unroll
  for (int i = 0; i < 4; ++i) BF[i] = bfi[i];

  const float PI_F = 3.14159265358979323846f;

  for (int tile = blockIdx.x * WAVES + wv; tile < nTiles; tile += gridDim.x * WAVES) {
    int p = tile * 16 + ln;
    int pc = p < nPoints ? p : nPoints - 1;

    const float4* xp = (const float4*)(x + (size_t)pc * 12);
    float4 q0 = xp[0], q1 = xp[1], q2 = xp[2];
    float u = q0.x, v = q0.y, fi = q0.z;

    // ---- parametrize (f32) ----
    F3 nn{q0.w, q1.x, q1.y}, wo{q1.z, q1.w, q2.x}, wi{q2.y, q2.z, q2.w};
    bool valid = (nn.x != 0.f) || (nn.y != 0.f) || (nn.z != 0.f);
    nn = nrm3(nn); wo = nrm3(wo); wi = nrm3(wi);
    F3 mid = nrm3(F3{nn.x, nn.y, nn.z + 1.f});
    float dwo = 2.f * (mid.x * wo.x + mid.y * wo.y + mid.z * wo.z);
    float dwi = 2.f * (mid.x * wi.x + mid.y * wi.y + mid.z * wi.z);
    F3 wo2{mid.x * dwo - wo.x, mid.y * dwo - wo.y, mid.z * dwo - wo.z};
    F3 wi2{mid.x * dwi - wi.x, mid.y * dwi - wi.y, mid.z * dwi - wi.z};
    F3 H = nrm3(F3{wo2.x + wi2.x, wo2.y + wi2.y, wo2.z + wi2.z});
    float cth = fminf(fmaxf(H.z, -1.f), 1.f);
    float th_h = acosf(cth);
    float sth = sqrtf(fmaxf(0.f, 1.f - cth * cth));
    float rho = sqrtf(H.x * H.x + H.y * H.y);
    float inv = rho > 1e-20f ? 1.f / rho : 0.f;
    float cph = rho > 1e-20f ? H.x * inv : 1.f;
    float sph = H.y * inv;
    F3 tmp = nrm3(F3{wi2.x * cph + wi2.y * sph, wi2.y * cph - wi2.x * sph, wi2.z});
    F3 dif = nrm3(F3{tmp.x * cth - tmp.z * sth, tmp.y, tmp.z * cth + tmp.x * sth});
    float th_d = acosf(fminf(fmaxf(dif.z, 0.f), 1.f));
    float ph_d = fmodf(atan2f(dif.y, dif.x), PI_F);
    float a0 = valid ? ph_d : 0.f;
    float a1 = valid ? th_h : 0.f;
    float a2 = valid ? th_d : 0.f;

    // ---- ang fourier features -> act0 slots 0..10 (zeros to 31) ----
    if (lh == 0) {
      union { _Float16 e[32]; half8 h4[4]; } F;
#pragma unroll
      for (int i = 0; i < 32; ++i) F.e[i] = (_Float16)0.f;
      F.e[0] = (_Float16)a0; F.e[1] = (_Float16)a1; F.e[2] = (_Float16)a2;
#pragma unroll
      for (int j = 0; j < 4; ++j) {
        float pr = BA[3 * j] * a0 + BA[3 * j + 1] * a1 + BA[3 * j + 2] * a2;
        float s, c; __sincosf(pr, &s, &c);
        F.e[3 + 2 * j] = (_Float16)s; F.e[4 + 2 * j] = (_Float16)c;
      }
      half8* dp = (half8*)(act0 + ln * ROW_B);
      dp[0] = F.h4[0]; dp[1] = F.h4[1]; dp[2] = F.h4[2]; dp[3] = F.h4[3];
    }

    // ---- ang MLP chain (wave-local, ping-pong LDS buffers, fully unrolled) ----
    float angY[3];
    {
      char* cur = act0; char* nxt = act1;
#pragma unroll
      for (int li = 26; li < 31; ++li) {
        run_layer(kT.L[li], smem, cur, nxt, ln, lh);
        char* t = cur; cur = nxt; nxt = t;
      }
      run_final(kT.L[31], smem, cur, ln, lh, angY);
    }

    // ---- uv/fi fourier features c[27] -> both buffers, zero slots 48..63 ----
    if (lh == 0) {
      union { _Float16 e[32]; half8 h4[4]; } F;
#pragma unroll
      for (int i = 0; i < 32; ++i) F.e[i] = (_Float16)0.f;
      F.e[0] = (_Float16)u; F.e[1] = (_Float16)v;
#pragma unroll
      for (int j = 0; j < 8; ++j) {
        float pr = BU[2 * j] * u + BU[2 * j + 1] * v;
        float s, c; __sincosf(pr, &s, &c);
        F.e[2 + 2 * j] = (_Float16)s; F.e[3 + 2 * j] = (_Float16)c;
      }
      F.e[18] = (_Float16)fi;
#pragma unroll
      for (int j = 0; j < 4; ++j) {
        float pr = BF[j] * fi;
        float s, c; __sincosf(pr, &s, &c);
        F.e[19 + 2 * j] = (_Float16)s; F.e[20 + 2 * j] = (_Float16)c;
      }
      half8 z = {};
      half8* d0 = (half8*)(act0 + ln * ROW_B);
      half8* d1 = (half8*)(act1 + ln * ROW_B);
      d0[0] = F.h4[0]; d0[1] = F.h4[1]; d0[2] = F.h4[2]; d0[3] = F.h4[3];
      d1[0] = F.h4[0]; d1[1] = F.h4[1]; d1[2] = F.h4[2]; d1[3] = F.h4[3];
      d0[6] = z; d0[7] = z;   // slots 48..63
      d1[6] = z; d1[7] = z;
    }

    // ---- uv MLP chain (fully unrolled) ----
    float uvY[3];
    {
      char* cur = act0; char* nxt = act1;
#pragma unroll
      for (int li = 0; li < 25; ++li) {
        run_layer(kT.L[li], smem, cur, nxt, ln, lh);
        char* t = cur; cur = nxt; nxt = t;
      }
      run_final(kT.L[25], smem, cur, ln, lh, uvY);
    }

    // ---- combine + store ----
    if (lh == 0 && p < nPoints) {
      float* op = out + (size_t)p * 3;
      op[0] = uvY[0] * angY[0];
      op[1] = uvY[1] * angY[1];
      op[2] = uvY[2] * angY[2];
    }
  }
}

// ---------------------------------------------------------------------------
// Host launcher
// ---------------------------------------------------------------------------
extern "C" void kernel_launch(void* const* d_in, const int* in_sizes, int n_in,
                              void* d_out, int out_size, void* d_ws, size_t ws_size,
                              hipStream_t stream) {
  if (n_in < 8) return;
  const float* x    = (const float*)d_in[0];
  const float* uvW  = (const float*)d_in[1];
  const float* uvB  = (const float*)d_in[2];
  const float* angW = (const float*)d_in[3];
  const float* angB = (const float*)d_in[4];
  const float* bu   = (const float*)d_in[5];
  const float* bfi  = (const float*)d_in[6];
  const float* ban  = (const float*)d_in[7];
  float* out = (float*)d_out;

  _Float16* wStage = (_Float16*)d_ws;
  float* bStage = (float*)((char*)d_ws + W_BYTES);

  bsdf_prep<<<64, 256, 0, stream>>>(uvW, uvB, angW, angB, wStage, bStage);

  int nPoints = in_sizes[0] / 12;
  int nTiles = (nPoints + 15) / 16;
  int blocks = 1024;
  if (blocks * WAVES > nTiles) blocks = (nTiles + WAVES - 1) / WAVES;
  if (blocks < 1) blocks = 1;

  bsdf_main<<<blocks, 256, SMEM_B, stream>>>(x, (const uint4*)d_ws, bu, bfi, ban,
                                             out, nPoints, nTiles);
}
